// ProbabilisticRoutingTree_71614284693841
// MI455X (gfx1250) — matrix operand order repaired and checked by reading, not compiled, not run on hardware
//
#include <hip/hip_runtime.h>

typedef __bf16 bf16;
typedef __attribute__((ext_vector_type(16))) __bf16 v16bf;
typedef __attribute__((ext_vector_type(8)))  float  v8f;
typedef int v4i_ __attribute__((vector_size(16)));

#define B_DIM    8192
#define D_DIM    1024
#define O_DIM    1024
#define HR_DIM   32
#define HE_DIM   2048
#define N_NODES  7
#define N_LEAVES 8

__device__ __forceinline__ float gelu_exact(float x) {
    return 0.5f * x * (1.0f + erff(x * 0.70710678118654752f));
}

// Async DMA of 16 bytes global -> LDS (GLOBAL_LOAD_ASYNC_TO_LDS_B128, ASYNCcnt).
__device__ __forceinline__ void async_copy16(const bf16* g, bf16* l) {
    __builtin_amdgcn_global_load_async_to_lds_b128(
        (__attribute__((address_space(1))) v4i_*)(unsigned long long)(const void*)g,
        (__attribute__((address_space(3))) v4i_*)(unsigned)(unsigned long long)(void*)l,
        0, 0);
}

// ---------------------------------------------------------------------------
// fp32 -> bf16 convert (grid-stride)
// ---------------------------------------------------------------------------
__global__ void convert_bf16_kernel(const float* __restrict__ in, bf16* __restrict__ out, size_t n) {
    size_t i = (size_t)blockIdx.x * blockDim.x + threadIdx.x;
    size_t stride = (size_t)gridDim.x * blockDim.x;
    for (; i < n; i += stride) out[i] = (bf16)in[i];
}

// ---------------------------------------------------------------------------
// fp32 [E][K][N] -> bf16 [E][N][K] tiled transpose (32x32 via LDS)
// ---------------------------------------------------------------------------
__global__ void transpose_bf16_kernel(const float* __restrict__ in, bf16* __restrict__ out,
                                      int K, int N) {
    __shared__ float tile[32][33];
    int e  = blockIdx.z;
    int k0 = blockIdx.x * 32, n0 = blockIdx.y * 32;
    int tx = threadIdx.x, ty = threadIdx.y;          // (32, 8)
    const float* src = in  + (size_t)e * K * N;
    bf16*        dst = out + (size_t)e * N * K;
    for (int i = 0; i < 32; i += 8)
        tile[ty + i][tx] = src[(size_t)(k0 + ty + i) * N + (n0 + tx)];
    __syncthreads();
    for (int i = 0; i < 32; i += 8)
        dst[(size_t)(n0 + ty + i) * K + (k0 + tx)] = (bf16)tile[tx][ty + i];
}

// ---------------------------------------------------------------------------
// Router: one wave computes 16 rows for one decision node via two
// 16x16x32 bf16 WMMA accumulators, then 32->2 logits + softmax through LDS.
// dec layout: [node][B][2] fp32.
// ---------------------------------------------------------------------------
__global__ __launch_bounds__(32) void router_kernel(
    const bf16* __restrict__ xb, const bf16* __restrict__ r1t,
    const float* __restrict__ rb1, const float* __restrict__ rW2, const float* __restrict__ rb2,
    float* __restrict__ dec) {
    __shared__ float hbuf[16][33];
    int lane = threadIdx.x;
    int node = blockIdx.y;
    int m0   = blockIdx.x * 16;
    int lrow = lane & 15, lhi = lane >> 4;
    v8f acc0 = {}, acc1 = {};
    const bf16* Arow = xb  + (size_t)(m0 + lrow) * D_DIM;
    const bf16* B0   = r1t + (size_t)(node * HR_DIM + lrow) * D_DIM;
    const bf16* B1   = B0 + (size_t)16 * D_DIM;
    for (int kb = 0; kb < D_DIM; kb += 32) {
        v16bf a, b0, b1;
        #pragma unroll
        for (int v = 0; v < 8; ++v) {
            int ka = (lhi << 4) + (v << 1);
            a[2*v]   = Arow[kb + ka];
            a[2*v+1] = Arow[kb + ka + 1];
            int kw = (lhi << 4) + (v << 1);
            b0[2*v]   = B0[kb + kw];
            b0[2*v+1] = B0[kb + kw + 1];
            b1[2*v]   = B1[kb + kw];
            b1[2*v+1] = B1[kb + kw + 1];
        }
        acc0 = __builtin_amdgcn_wmma_f32_16x16x32_bf16(false, a, false, b0, (short)0, acc0, false, false);
        acc1 = __builtin_amdgcn_wmma_f32_16x16x32_bf16(false, a, false, b1, (short)0, acc1, false, false);
    }
    #pragma unroll
    for (int r = 0; r < 8; ++r) {
        int m = lhi * 8 + r;
        hbuf[m][lrow]      = gelu_exact(acc0[r] + rb1[node * HR_DIM + lrow]);
        hbuf[m][16 + lrow] = gelu_exact(acc1[r] + rb1[node * HR_DIM + 16 + lrow]);
    }
    __syncthreads();
    if (lane < 16) {
        int m = lane;
        float l0 = rb2[node * 2], l1 = rb2[node * 2 + 1];
        for (int j = 0; j < HR_DIM; ++j) {
            float hv = hbuf[m][j];
            l0 += hv * rW2[(node * HR_DIM + j) * 2];
            l1 += hv * rW2[(node * HR_DIM + j) * 2 + 1];
        }
        float mx = fmaxf(l0, l1);
        float e0 = expf(l0 - mx), e1 = expf(l1 - mx);
        float inv = 1.0f / (e0 + e1);
        size_t o = ((size_t)node * B_DIM + (size_t)(m0 + m)) * 2;
        dec[o]     = e0 * inv;
        dec[o + 1] = e1 * inv;
    }
}

// ---------------------------------------------------------------------------
// Leaf probabilities (BFS path product) + top-k sparsify + renormalize.
// ---------------------------------------------------------------------------
__global__ void leaf_topk_kernel(const float* __restrict__ dec, const int* __restrict__ topk,
                                 float* __restrict__ lp) {
    int b = blockIdx.x * blockDim.x + threadIdx.x;
    if (b >= B_DIM) return;
    float dl[N_NODES][2];
    #pragma unroll
    for (int n = 0; n < N_NODES; ++n) {
        size_t o = ((size_t)n * B_DIM + b) * 2;
        dl[n][0] = dec[o]; dl[n][1] = dec[o + 1];
    }
    float leaf[N_LEAVES];
    #pragma unroll
    for (int i = 0; i < N_LEAVES; ++i) {
        int c0 = (i >> 2) & 1, c1 = (i >> 1) & 1, c2 = i & 1;
        int n1 = 1 + c0;
        int n2 = 3 + (i >> 1);
        leaf[i] = dl[0][c0] * dl[n1][c1] * dl[n2][c2];
    }
    int k = *topk;
    if (k > 0 && k < N_LEAVES) {
        float kept[N_LEAVES];
        bool  sel[N_LEAVES];
        #pragma unroll
        for (int i = 0; i < N_LEAVES; ++i) { sel[i] = false; kept[i] = 0.0f; }
        for (int t = 0; t < k; ++t) {
            int best = 0; float bv = -1.0f;
            #pragma unroll
            for (int i = 0; i < N_LEAVES; ++i)
                if (!sel[i] && leaf[i] > bv) { bv = leaf[i]; best = i; }
            sel[best] = true; kept[best] = leaf[best];
        }
        float s = 1e-8f;
        #pragma unroll
        for (int i = 0; i < N_LEAVES; ++i) s += kept[i];
        float inv = 1.0f / s;
        #pragma unroll
        for (int i = 0; i < N_LEAVES; ++i) leaf[i] = kept[i] * inv;
    }
    #pragma unroll
    for (int i = 0; i < N_LEAVES; ++i) lp[(size_t)b * N_LEAVES + i] = leaf[i];
}

// ---------------------------------------------------------------------------
// Tiled bf16 WMMA GEMM with double-buffered ASYNC global->LDS tile staging.
//   C = A[MxK] * Wt[NxK]^T (+bias), epilogues:
//   MODE 0: Hout = bf16(gelu(C + bias))                  (expert layer 1)
//   MODE 1: Y (+)= P[m, expert] * (C + bias)             (expert layer 2)
// Block: 256 threads = 8 waves (2 M-waves x 4 N-waves), block tile 128x128,
// wave tile 64x32 = 4x2 WMMA accumulators, K-step 32.
// Each thread issues 4 x 16B GLOBAL_LOAD_ASYNC_TO_LDS_B128 per stage;
// s_wait_asynccnt(4) retires the current stage while the next is in flight.
// ---------------------------------------------------------------------------
template <int MODE>
__global__ __launch_bounds__(256) void gemm_bf16_kernel(
    const bf16* __restrict__ A, const bf16* __restrict__ Wt, const float* __restrict__ bias,
    bf16* __restrict__ Hout, const float* __restrict__ P, float* __restrict__ Y,
    int expert, int first, int M, int N, int K) {
    __shared__ bf16 As[2][128][32];
    __shared__ bf16 Bs[2][128][32];
    int tid  = threadIdx.x;
    int lane = tid & 31, wave = tid >> 5;
    int wr   = wave >> 2, wc = wave & 3;     // 2 x 4 wave grid
    int lrow = lane & 15, lhi = lane >> 4;
    size_t m0 = (size_t)blockIdx.x * 128;
    size_t n0 = (size_t)blockIdx.y * 128;

    // this thread's two 16B chunks within the 128x32 tile
    int row0 = tid >> 2, seg0 = (tid & 3) << 3;
    int row1 = (tid + 256) >> 2, seg1 = ((tid + 256) & 3) << 3;

    v8f acc[4][2];
    #pragma unroll
    for (int mi = 0; mi < 4; ++mi)
        #pragma unroll
        for (int ni = 0; ni < 2; ++ni)
            acc[mi][ni] = {};

    const int nk = K >> 5;  // K / 32 stages

    // prologue: stage 0 into buffer 0
    async_copy16(A  + (m0 + row0) * K + seg0, &As[0][row0][seg0]);
    async_copy16(A  + (m0 + row1) * K + seg1, &As[0][row1][seg1]);
    async_copy16(Wt + (n0 + row0) * K + seg0, &Bs[0][row0][seg0]);
    async_copy16(Wt + (n0 + row1) * K + seg1, &Bs[0][row1][seg1]);

    for (int i = 0; i < nk; ++i) {
        int cur = i & 1;
        if (i + 1 < nk) {
            int nxt = cur ^ 1;
            int kb = (i + 1) << 5;
            async_copy16(A  + (m0 + row0) * K + kb + seg0, &As[nxt][row0][seg0]);
            async_copy16(A  + (m0 + row1) * K + kb + seg1, &As[nxt][row1][seg1]);
            async_copy16(Wt + (n0 + row0) * K + kb + seg0, &Bs[nxt][row0][seg0]);
            async_copy16(Wt + (n0 + row1) * K + kb + seg1, &Bs[nxt][row1][seg1]);
            __builtin_amdgcn_s_wait_asynccnt(4);   // stage i complete, stage i+1 in flight
        } else {
            __builtin_amdgcn_s_wait_asynccnt(0);   // last stage complete
        }
        __syncthreads();

        v16bf af[4], bfr[2];
        #pragma unroll
        for (int mi = 0; mi < 4; ++mi) {
            int row = wr * 64 + mi * 16 + lrow;
            #pragma unroll
            for (int v = 0; v < 8; ++v) {
                int k0 = (lhi << 4) + (v << 1);
                af[mi][2*v]   = As[cur][row][k0];
                af[mi][2*v+1] = As[cur][row][k0 + 1];
            }
        }
        #pragma unroll
        for (int ni = 0; ni < 2; ++ni) {
            int col = wc * 32 + ni * 16 + lrow;
            #pragma unroll
            for (int v = 0; v < 8; ++v) {
                int k0 = (lhi << 4) + (v << 1);
                bfr[ni][2*v]   = Bs[cur][col][k0];
                bfr[ni][2*v+1] = Bs[cur][col][k0 + 1];
            }
        }
        #pragma unroll
        for (int mi = 0; mi < 4; ++mi)
            #pragma unroll
            for (int ni = 0; ni < 2; ++ni)
                acc[mi][ni] = __builtin_amdgcn_wmma_f32_16x16x32_bf16(
                    false, af[mi], false, bfr[ni], (short)0, acc[mi][ni], false, false);
        __syncthreads();   // protect buffer 'cur' from being overwritten next iteration
    }

    #pragma unroll
    for (int mi = 0; mi < 4; ++mi) {
        #pragma unroll
        for (int ni = 0; ni < 2; ++ni) {
            size_t ncol = n0 + (size_t)(wc * 32 + ni * 16 + lrow);
            float bv = bias[ncol];
            #pragma unroll
            for (int r = 0; r < 8; ++r) {
                size_t mrow = m0 + (size_t)(wr * 64 + mi * 16 + lhi * 8 + r);
                float v = acc[mi][ni][r] + bv;
                if (MODE == 0) {
                    Hout[mrow * N + ncol] = (bf16)gelu_exact(v);
                } else {
                    float pw  = P[mrow * N_LEAVES + expert];
                    float val = pw * v;
                    float* yp = Y + mrow * N + ncol;
                    if (first) *yp = val;
                    else       *yp += val;
                }
            }
        }
    }
}

// ---------------------------------------------------------------------------
extern "C" void kernel_launch(void* const* d_in, const int* in_sizes, int n_in,
                              void* d_out, int out_size, void* d_ws, size_t ws_size,
                              hipStream_t stream) {
    const float* x   = (const float*)d_in[0];
    const float* rW1 = (const float*)d_in[1];
    const float* rb1 = (const float*)d_in[2];
    const float* rW2 = (const float*)d_in[3];
    const float* rb2 = (const float*)d_in[4];
    const float* eW1 = (const float*)d_in[5];
    const float* eb1 = (const float*)d_in[6];
    const float* eW2 = (const float*)d_in[7];
    const float* eb2 = (const float*)d_in[8];
    const int*  topk = (const int*)d_in[9];

    float* Y  = (float*)d_out;
    float* LP = Y + (size_t)B_DIM * O_DIM;   // leaf_probs output region [B, 8]

    char* p = (char*)d_ws;
    auto take = [&](size_t bytes) { char* q = p; p += (bytes + 255) & ~(size_t)255; return q; };
    bf16*  xb   = (bf16*) take((size_t)B_DIM * D_DIM * 2);
    bf16*  e1t  = (bf16*) take((size_t)N_LEAVES * HE_DIM * D_DIM * 2);   // [E][HE][D]
    bf16*  e2t  = (bf16*) take((size_t)N_LEAVES * O_DIM * HE_DIM * 2);   // [E][O][HE]
    bf16*  r1t  = (bf16*) take((size_t)N_NODES * HR_DIM * D_DIM * 2);    // [N][HR][D]
    float* dec  = (float*)take((size_t)N_NODES * B_DIM * 2 * sizeof(float));
    bf16*  Hbuf = (bf16*) take((size_t)B_DIM * HE_DIM * 2);

    convert_bf16_kernel<<<8192, 256, 0, stream>>>(x, xb, (size_t)B_DIM * D_DIM);
    transpose_bf16_kernel<<<dim3(D_DIM / 32, HR_DIM / 32, N_NODES), dim3(32, 8), 0, stream>>>(
        rW1, r1t, D_DIM, HR_DIM);
    transpose_bf16_kernel<<<dim3(D_DIM / 32, HE_DIM / 32, N_LEAVES), dim3(32, 8), 0, stream>>>(
        eW1, e1t, D_DIM, HE_DIM);
    transpose_bf16_kernel<<<dim3(HE_DIM / 32, O_DIM / 32, N_LEAVES), dim3(32, 8), 0, stream>>>(
        eW2, e2t, HE_DIM, O_DIM);

    router_kernel<<<dim3(B_DIM / 16, N_NODES), 32, 0, stream>>>(xb, r1t, rb1, rW2, rb2, dec);
    leaf_topk_kernel<<<B_DIM / 256, 256, 0, stream>>>(dec, topk, LP);

    for (int e = 0; e < N_LEAVES; ++e) {
        gemm_bf16_kernel<0><<<dim3(B_DIM / 128, HE_DIM / 128), 256, 0, stream>>>(
            xb, e1t + (size_t)e * HE_DIM * D_DIM, eb1 + (size_t)e * HE_DIM,
            Hbuf, nullptr, nullptr, 0, 0, B_DIM, HE_DIM, D_DIM);
        gemm_bf16_kernel<1><<<dim3(B_DIM / 128, O_DIM / 128), 256, 0, stream>>>(
            Hbuf, e2t + (size_t)e * O_DIM * HE_DIM, eb2 + (size_t)e * O_DIM,
            nullptr, LP, Y, e, (e == 0) ? 1 : 0, B_DIM, O_DIM, HE_DIM);
    }
    (void)in_sizes; (void)n_in; (void)out_size; (void)ws_size;
}
